// NbrAttnGapFill_29532195127724
// MI455X (gfx1250) — matrix operand order repaired and checked by reading, not compiled, not run on hardware
//
#include <hip/hip_runtime.h>
#include <math.h>

// ---------------------------------------------------------------------------
// CDNA5 (gfx1250) implementation of NbrAttnGapFill forward.
// All GEMMs run through v_wmma_f32_16x16x32_f16 (wave32, 32x16 C tile/wave).
// ---------------------------------------------------------------------------

typedef __attribute__((ext_vector_type(16))) _Float16 v16h;
typedef __attribute__((ext_vector_type(8)))  float    v8f;

// ------------------------------- GEMM --------------------------------------
// C[b,h] = act( A[b,h] (MxK) @ op(B[b,h]) + bias + R[b,h] )
//   transB=1: B is (N,K) row-major (torch Linear weight, y = x @ W^T)
//   transB=0: B is (K,N) row-major (plain A @ B, e.g. attn @ V)
// Workgroup: 256 thr = 8 waves. WG tile 128(M) x 32(N); wave tile 32x16
// (2 accumulators sharing one B fragment). K stepped by 32 via LDS f16 tiles.
// FA: M%128==0 && lda%4==0 && K%32==0  -> unconditional float4 staging of A
// FB: N%32==0  && ldb%4==0 && K%32==0  -> unconditional float4 staging of B
// Variant chosen on host, so the fast inner loop has zero guard logic.
template <bool FA, bool FB>
__global__ __launch_bounds__(256)
void k_gemm_t(const float* __restrict__ A, long long sAb, long long sAh, int lda,
              const float* __restrict__ B, long long sBb, long long sBh, int ldb, int transB,
              const float* __restrict__ bias,
              const float* __restrict__ R, long long sRb, long long sRh,
              float* __restrict__ C, long long sCb, long long sCh, int ldc,
              int M, int N, int K, int batchH, int act)
{
  __shared__ __align__(16) _Float16 lA[128][36];
  __shared__ __align__(16) _Float16 lB[32][36];

  const int tid  = threadIdx.x;
  const int lane = tid & 31;
  const int wid  = tid >> 5;
  const int wm   = wid >> 1;          // 0..3  -> rows wm*32
  const int wn   = wid & 1;           // 0..1  -> cols wn*16
  const int m0   = blockIdx.x * 128;
  const int n0   = blockIdx.y * 32;
  const int bz   = blockIdx.z;
  const int b    = bz / batchH;
  const int h    = bz - b * batchH;

  const float* Ab = A + (long long)b * sAb + (long long)h * sAh;
  const float* Bb = B + (long long)b * sBb + (long long)h * sBh;

  v8f acc0 = {0.f, 0.f, 0.f, 0.f, 0.f, 0.f, 0.f, 0.f};
  v8f acc1 = {0.f, 0.f, 0.f, 0.f, 0.f, 0.f, 0.f, 0.f};

  const int khalf = lane >> 4;        // 0 or 1
  const int l16   = lane & 15;

  for (int k0 = 0; k0 < K; k0 += 32) {
    // ---- stage A tile (128 x 32) -----------------------------------------
    if constexpr (FA) {
      #pragma unroll
      for (int q = 0; q < 4; ++q) {
        int quad = tid + 256 * q;               // 1024 float4 quads
        int r  = quad >> 3;                     // 8 quads per row
        int kk = (quad & 7) << 2;
        float4 v = *(const float4*)(Ab + (long long)(m0 + r) * lda + k0 + kk);
        union { _Float16 h[4]; uint2 u; } pk;
        pk.h[0] = (_Float16)v.x; pk.h[1] = (_Float16)v.y;
        pk.h[2] = (_Float16)v.z; pk.h[3] = (_Float16)v.w;
        *(uint2*)&lA[r][kk] = pk.u;             // 8B aligned (stride 72B)
      }
    } else {
      // branch-free guarded path: gather all, then convert+store
      float va[16];
      bool  ok[16];
      #pragma unroll
      for (int q = 0; q < 16; ++q) {
        int i = tid + 256 * q;                  // 4096 elements
        int r = i >> 5, kk = i & 31;
        int gm = m0 + r, gk = k0 + kk;
        ok[q] = (gm < M) && (gk < K);
        long long off = ok[q] ? ((long long)gm * lda + gk) : 0;
        va[q] = Ab[off];
      }
      #pragma unroll
      for (int q = 0; q < 16; ++q) {
        int i = tid + 256 * q;
        int r = i >> 5, kk = i & 31;
        lA[r][kk] = (_Float16)(ok[q] ? va[q] : 0.f);
      }
    }

    // ---- stage B tile (32 n-rows x 32 k) ---------------------------------
    if constexpr (FB) {
      if (transB) {
        int r  = tid >> 3;                      // 256 quads: 32 rows x 8
        int kk = (tid & 7) << 2;
        float4 v = *(const float4*)(Bb + (long long)(n0 + r) * ldb + k0 + kk);
        union { _Float16 h[4]; uint2 u; } pk;
        pk.h[0] = (_Float16)v.x; pk.h[1] = (_Float16)v.y;
        pk.h[2] = (_Float16)v.z; pk.h[3] = (_Float16)v.w;
        *(uint2*)&lB[r][kk] = pk.u;
      } else {
        int kk = tid >> 3;                      // k row
        int nq = (tid & 7) << 2;                // 4 consecutive n
        float4 v = *(const float4*)(Bb + (long long)(k0 + kk) * ldb + n0 + nq);
        lB[nq + 0][kk] = (_Float16)v.x;
        lB[nq + 1][kk] = (_Float16)v.y;
        lB[nq + 2][kk] = (_Float16)v.z;
        lB[nq + 3][kk] = (_Float16)v.w;
      }
    } else {
      float vb[4];
      bool  ok[4];
      #pragma unroll
      for (int q = 0; q < 4; ++q) {
        int i = tid + 256 * q;                  // 1024 elements
        int nn = i >> 5, kk = i & 31;
        int gn = n0 + nn, gk = k0 + kk;
        ok[q] = (gn < N) && (gk < K);
        long long off = ok[q] ? (transB ? ((long long)gn * ldb + gk)
                                        : ((long long)gk * ldb + gn)) : 0;
        vb[q] = Bb[off];
      }
      #pragma unroll
      for (int q = 0; q < 4; ++q) {
        int i = tid + 256 * q;
        int nn = i >> 5, kk = i & 31;
        lB[nn][kk] = (_Float16)(ok[q] ? vb[q] : 0.f);
      }
    }
    __syncthreads();

    // ---- fragments per ISA 7.12.2 (wave32, 16-bit A 16x32 / B 32x16) -----
    union { v16h v; unsigned u[8]; } af0, af1, bf;
    const int am0 = wm * 32 + l16;
    #pragma unroll
    for (int j = 0; j < 8; ++j) {
      int grp = j >> 2, jj = j & 3;
      int k = grp * 16 + khalf * 8 + jj * 2;    // packed f16 pair (k, k+1)
      af0.u[j] = *(const unsigned*)&lA[am0][k];
      af1.u[j] = *(const unsigned*)&lA[am0 + 16][k];
    }
    const int bn = wn * 16 + l16;
    const int kb = khalf * 16;
    #pragma unroll
    for (int j = 0; j < 8; ++j)
      bf.u[j] = *(const unsigned*)&lB[bn][(j >> 2) * 16 + khalf * 8 + (j & 3) * 2];

    acc0 = __builtin_amdgcn_wmma_f32_16x16x32_f16(false, af0.v, false, bf.v,
                                                  (short)0, acc0, false, false);
    acc1 = __builtin_amdgcn_wmma_f32_16x16x32_f16(false, af1.v, false, bf.v,
                                                  (short)0, acc1, false, false);
    __syncthreads();
  }

  // ---- epilogue: D layout — lane holds n = lane%16, m = j + 8*(lane>=16) --
  const int gn  = n0 + wn * 16 + l16;
  const bool nok = FB || (gn < N);
  const float bv = (bias != nullptr && nok) ? bias[gn] : 0.f;
  const float* Rb = R ? (R + (long long)b * sRb + (long long)h * sRh) : nullptr;
  float* Cb = C + (long long)b * sCb + (long long)h * sCh;
  #pragma unroll
  for (int j = 0; j < 8; ++j) {
    int gm = m0 + wm * 32 + khalf * 8 + j;
    #pragma unroll
    for (int half = 0; half < 2; ++half) {
      int gmm = gm + 16 * half;
      bool mok = FA || (gmm < M);
      if (mok && nok) {
        float v = (half ? acc1[j] : acc0[j]) + bv;
        if (Rb) v += Rb[(long long)gmm * ldc + gn];
        if (act == 1)      v = fmaxf(v, 0.f);
        else if (act == 2) v = 0.5f * v * (1.f + erff(v * 0.70710678118654752f));
        Cb[(long long)gmm * ldc + gn] = v;
      }
    }
  }
}

// --------------------------- masked softmax --------------------------------
// One row per workgroup. val = S*scale; mask==0 -> -1e9 (matches reference).
// mask layout (B, Nn, Lq, Tn) int32; key index j -> (n = j/Tn, t = j%Tn).
__global__ __launch_bounds__(256)
void k_softmax(float* __restrict__ S, int Lq, int Lk, float scale,
               const int* __restrict__ mask, int Nn, int Tn, int batchH)
{
  const int q  = blockIdx.x;
  const int bz = blockIdx.y;
  float* row = S + ((long long)bz * Lq + q) * (long long)Lk;
  const int b   = bz / batchH;
  const int tid = threadIdx.x;
  __shared__ float red[256];

  float mx = -3.4e38f;
  for (int j = tid; j < Lk; j += 256) {
    float v = row[j] * scale;
    if (mask) {
      int n = j / Tn, t = j - n * Tn;
      if (mask[(((long long)b * Nn + n) * Lq + q) * Tn + t] == 0) v = -1e9f;
    }
    row[j] = v;
    mx = fmaxf(mx, v);
  }
  red[tid] = mx; __syncthreads();
  for (int s = 128; s > 0; s >>= 1) { if (tid < s) red[tid] = fmaxf(red[tid], red[tid + s]); __syncthreads(); }
  mx = red[0]; __syncthreads();

  float sum = 0.f;
  for (int j = tid; j < Lk; j += 256) {
    float e = __expf(row[j] - mx);
    row[j] = e;
    sum += e;
  }
  red[tid] = sum; __syncthreads();
  for (int s = 128; s > 0; s >>= 1) { if (tid < s) red[tid] += red[tid + s]; __syncthreads(); }
  const float inv = 1.f / red[0];
  for (int j = tid; j < Lk; j += 256) row[j] *= inv;
}

// ------------------------------ layernorm ----------------------------------
// Reference: a*(x-m)/(std+eps)+b with unbiased std (ddof=1), eps on std.
__global__ __launch_bounds__(256)
void k_layernorm(const float* __restrict__ X, const float* __restrict__ g,
                 const float* __restrict__ be, float* __restrict__ Y, int d)
{
  const long long row = blockIdx.x;
  const float* x = X + row * d;
  float* y = Y + row * d;
  const int tid = threadIdx.x;
  __shared__ float red[256];

  float s = 0.f;
  for (int c = tid; c < d; c += 256) s += x[c];
  red[tid] = s; __syncthreads();
  for (int k = 128; k > 0; k >>= 1) { if (tid < k) red[tid] += red[tid + k]; __syncthreads(); }
  const float mean = red[0] / d; __syncthreads();

  float ss = 0.f;
  for (int c = tid; c < d; c += 256) { float t = x[c] - mean; ss += t * t; }
  red[tid] = ss; __syncthreads();
  for (int k = 128; k > 0; k >>= 1) { if (tid < k) red[tid] += red[tid + k]; __syncthreads(); }
  const float sd  = sqrtf(red[0] / (float)(d - 1));
  const float inv = 1.f / (sd + 1e-6f);
  for (int c = tid; c < d; c += 256) y[c] = g[c] * (x[c] - mean) * inv + be[c];
}

// -------------------- fused PE add (+ optional product) --------------------
// dst[r, c] = (a * (b?)) + PE(t = r % T, c);  nld = -ln(10000)/d
__global__ void k_addpe(float* __restrict__ dst, const float* __restrict__ a,
                        const float* __restrict__ b, int T, int d, float nld,
                        long long total)
{
  long long idx = (long long)blockIdx.x * blockDim.x + threadIdx.x;
  if (idx >= total) return;
  long long r = idx / d;
  int c = (int)(idx - r * d);
  int t = (int)(r % T);
  float freq = expf((float)(c & ~1) * nld);
  float ang  = (float)t * freq;
  float pe   = (c & 1) ? cosf(ang) : sinf(ang);
  float v = a[idx];
  if (b) v *= b[idx];
  dst[idx] = v + pe;
}

// --------------------------- transposes ------------------------------------
// (batch, C, R) channel-major -> (batch, R, ldc) at column offset `off`
__global__ void k_bct2btc(const float* __restrict__ in, float* __restrict__ out,
                          int C, int R, int off, int ldc, long long total)
{
  long long idx = (long long)blockIdx.x * blockDim.x + threadIdx.x;
  if (idx >= total) return;
  long long b = idx / ((long long)C * R);
  long long rem = idx - b * (long long)C * R;
  int c = (int)(rem / R);
  int r = (int)(rem - (long long)c * R);
  out[(b * R + r) * (long long)ldc + off + c] = in[idx];
}

// (batch, R, C) -> (batch, C, R) contiguous output
__global__ void k_btc2bct(const float* __restrict__ in, float* __restrict__ out,
                          int R, int C, long long total)
{
  long long idx = (long long)blockIdx.x * blockDim.x + threadIdx.x;
  if (idx >= total) return;
  long long b = idx / ((long long)C * R);
  long long rem = idx - b * (long long)C * R;
  int c = (int)(rem / R);
  int r = (int)(rem - (long long)c * R);
  out[idx] = in[(b * R + r) * (long long)C + c];
}

// ------------------------------ host side ----------------------------------
namespace {

constexpr int Bn = 16, Tt = 512, Nn = 4;
constexpr int FD = 8, EDS = 192, FAUX = 16, EAUX = 64;
constexpr int Dm = 256, EAC = 8, PEMB = 64;
constexpr int Hh = 8, HN = 4, NL = 4, DFF = 1024;

inline void gemm(hipStream_t st,
                 const float* A, long long sAb, long long sAh, int lda,
                 const float* B, long long sBb, long long sBh, int ldb, int transB,
                 const float* bias, const float* R, long long sRb, long long sRh,
                 float* C, long long sCb, long long sCh, int ldc,
                 int M, int N, int K, int batches, int batchH, int act)
{
  dim3 grid((M + 127) / 128, (N + 31) / 32, batches);
  const bool fa = ((M & 127) == 0) && ((lda & 3) == 0) && ((K & 31) == 0);
  const bool fb = ((N & 31) == 0) && ((ldb & 3) == 0) && ((K & 31) == 0);
  if (fa && fb)
    k_gemm_t<true, true><<<grid, dim3(256), 0, st>>>(A, sAb, sAh, lda, B, sBb, sBh, ldb,
        transB, bias, R, sRb, sRh, C, sCb, sCh, ldc, M, N, K, batchH, act);
  else if (fa)
    k_gemm_t<true, false><<<grid, dim3(256), 0, st>>>(A, sAb, sAh, lda, B, sBb, sBh, ldb,
        transB, bias, R, sRb, sRh, C, sCb, sCh, ldc, M, N, K, batchH, act);
  else if (fb)
    k_gemm_t<false, true><<<grid, dim3(256), 0, st>>>(A, sAb, sAh, lda, B, sBb, sBh, ldb,
        transB, bias, R, sRb, sRh, C, sCb, sCh, ldc, M, N, K, batchH, act);
  else
    k_gemm_t<false, false><<<grid, dim3(256), 0, st>>>(A, sAb, sAh, lda, B, sBb, sBh, ldb,
        transB, bias, R, sRb, sRh, C, sCb, sCh, ldc, M, N, K, batchH, act);
}

// plain linear: C = act(A (MxK) @ W^T (N,K) + b [+ R]) — unbatched
inline void lin(hipStream_t st, const float* A, int lda, const float* W,
                const float* b, float* C, int ldc, int M, int N, int K, int act,
                const float* R = nullptr)
{
  gemm(st, A, 0, 0, lda, W, 0, 0, K, 1, b, R, 0, 0, C, 0, 0, ldc, M, N, K, 1, 1, act);
}

inline void ln(hipStream_t st, const float* X, const float* g, const float* b,
               float* Y, int rows, int d)
{
  k_layernorm<<<dim3(rows), dim3(256), 0, st>>>(X, g, b, Y, d);
}

inline void addpe(hipStream_t st, float* dst, const float* a, const float* b,
                  int T, int d, long long total)
{
  float nld = -logf(10000.f) / (float)d;
  long long blocks = (total + 255) / 256;
  k_addpe<<<dim3((unsigned)blocks), dim3(256), 0, st>>>(dst, a, b, T, d, nld, total);
}

inline void bct2btc(hipStream_t st, const float* in, float* out, int C, int R,
                    int off, int ldc, long long total)
{
  long long blocks = (total + 255) / 256;
  k_bct2btc<<<dim3((unsigned)blocks), dim3(256), 0, st>>>(in, out, C, R, off, ldc, total);
}

} // namespace

extern "C" void kernel_launch(void* const* d_in, const int* in_sizes, int n_in,
                              void* d_out, int out_size, void* d_ws, size_t ws_size,
                              hipStream_t stream)
{
  (void)in_sizes; (void)n_in; (void)out_size; (void)ws_size;
  hipStream_t st = stream;

  // ---- inputs --------------------------------------------------------------
  const float* x_in   = (const float*)d_in[0];  // (B,8,T)
  const float* aux_in = (const float*)d_in[1];  // (B,16,T)
  const float* am_in  = (const float*)d_in[2];  // (B,N,8,T)
  const float* an_in  = (const float*)d_in[3];  // (B,N,16,T)
  const float* ae_in  = (const float*)d_in[4];  // (B,N,8,T)
  const int*   amask  = (const int*)  d_in[5];  // (B,N,T,T)
  const float* pm_in  = (const float*)d_in[6];  // (B,N,1,T)
  const float* pn_in  = (const float*)d_in[7];  // (B,N,16,T)
  const float* pe_in  = (const float*)d_in[8];  // (B,N,8,T)
  const int*   pmask  = (const int*)  d_in[9];  // (B,N,T,T)

  // params flattened depth-first in dict order, (w,b) per linear
  const float* P[134];
  for (int i = 10; i < 134; ++i) P[i] = (const float*)d_in[i];

  float* out = (float*)d_out;                   // [xa (B,3,T) | xp (B,1,T)]

  // ---- workspace plan (floats) --------------------------------------------
  float* W = (float*)d_ws;
  const long long S_SZ = 67108864LL;            // scores arena (256 MiB)
  float* S  = W;
  float* G0 = S  + S_SZ;                        // 4 x 8,388,608 (32 MiB each)
  float* G1 = G0 + 8388608LL;
  float* G2 = G1 + 8388608LL;
  float* G3 = G2 + 8388608LL;
  float* X  = G3 + 8388608LL;                   // (B,T,256) arenas, 8 MiB each
  float* XN = X  + 2097152LL;
  float* XF = XN + 2097152LL;
  float* T0 = XF + 2097152LL;
  float* T1 = T0 + 2097152LL;
  float* T2 = T1 + 2097152LL;                   // total ~432 MiB

  const int BT  = Bn * Tt;                      // 8192
  const int BNT = Bn * Nn * Tt;                 // 32768
  const int LK  = Nn * Tt;                      // 2048

  // =================== input embeds =======================================
  bct2btc(st, x_in, T0, FD, Tt, 0, FD, (long long)Bn * FD * Tt);
  lin(st, T0, FD, P[10], P[11], T1, EDS, BT, EDS, FD, 2);           // ie1 + gelu
  lin(st, T1, EDS, P[12], P[13], X, Dm, BT, EDS, EDS, 2);           // ie2 -> X[:, :192]
  bct2btc(st, aux_in, T0, FAUX, Tt, 0, FAUX, (long long)Bn * FAUX * Tt);
  lin(st, T0, FAUX, P[14], P[15], T1, EAUX, BT, EAUX, FAUX, 2);     // ae1 + gelu
  lin(st, T1, EAUX, P[16], P[17], X + EDS, Dm, BT, EAUX, EAUX, 2);  // ae2 -> X[:,192:]
  addpe(st, X, X, nullptr, Tt, Dm, (long long)BT * Dm);

  // =================== transformer layers =================================
  for (int l = 0; l < NL; ++l) {
    const int p = 18 + 16 * l;
    const int dk = Dm / Hh;                     // 32
    ln(st, X, P[p + 12], P[p + 13], XN, BT, Dm);
    lin(st, XN, Dm, P[p + 0], P[p + 1], T0, Dm, BT, Dm, Dm, 0);     // Q
    lin(st, XN, Dm, P[p + 2], P[p + 3], T1, Dm, BT, Dm, Dm, 0);     // K
    lin(st, XN, Dm, P[p + 4], P[p + 5], T2, Dm, BT, Dm, Dm, 0);     // V
    // scores (B,H,T,T) = Q @ K^T
    gemm(st, T0, (long long)Tt * Dm, dk, Dm,
             T1, (long long)Tt * Dm, dk, Dm, 1,
             nullptr, nullptr, 0, 0,
             S, (long long)Hh * Tt * Tt, (long long)Tt * Tt, Tt,
             Tt, Tt, dk, Bn * Hh, Hh, 0);
    k_softmax<<<dim3(Tt, Bn * Hh), dim3(256), 0, st>>>(S, Tt, Tt, 1.f / sqrtf((float)dk),
                                                       nullptr, 0, 1, Hh);
    // O = attn @ V  (heads merged via ldc/offset)
    gemm(st, S, (long long)Hh * Tt * Tt, (long long)Tt * Tt, Tt,
             T2, (long long)Tt * Dm, dk, Dm, 0,
             nullptr, nullptr, 0, 0,
             T0, (long long)Tt * Dm, dk, Dm,
             Tt, dk, Tt, Bn * Hh, Hh, 0);
    lin(st, T0, Dm, P[p + 6], P[p + 7], X, Dm, BT, Dm, Dm, 0, X);   // out proj + res
    ln(st, X, P[p + 14], P[p + 15], XN, BT, Dm);
    lin(st, XN, Dm, P[p + 8], P[p + 9], G0, DFF, BT, DFF, Dm, 1);   // ff1 + relu
    lin(st, G0, DFF, P[p + 10], P[p + 11], X, Dm, BT, Dm, DFF, 0, X); // ff2 + res
  }
  ln(st, X, P[82], P[83], XF, BT, Dm);          // final layernorm -> XF

  // =================== allvar neighbor-attention branch ===================
  {
    const int dk = Dm / HN;                     // 64
    // anbr = cat(ts, au) -> G0
    bct2btc(st, am_in, T0, FD, Tt, 0, FD, (long long)BNT * FD);
    lin(st, T0, FD, P[86], P[87], G1, EDS, BNT, EDS, FD, 1);        // ts1 + relu
    lin(st, G1, EDS, P[88], P[89], G0, Dm, BNT, EDS, EDS, 0);       // ts2 -> cols 0..191
    bct2btc(st, an_in, T0, FAUX, Tt, 0, FAUX, (long long)BNT * FAUX);
    lin(st, T0, FAUX, P[90], P[91], T1, EAUX, BNT, EAUX, FAUX, 1);  // na1 + relu
    lin(st, T1, EAUX, P[92], P[93], G0 + EDS, Dm, BNT, EAUX, EAUX, 0); // na2 -> cols 192..
    // edge -> G2
    bct2btc(st, ae_in, T0, EAC, Tt, 0, EAC, (long long)BNT * EAC);
    lin(st, T0, EAC, P[94], P[95], G1, Dm, BNT, Dm, EAC, 1);        // ea1 + relu
    lin(st, G1, Dm, P[96], P[97], G2, Dm, BNT, Dm, Dm, 0);          // ea2
    // keys/vals with PE
    addpe(st, G3, G0, G2, Tt, Dm, (long long)BNT * Dm);             // keys = nbr*edge+PE
    addpe(st, G1, G0, nullptr, Tt, Dm, (long long)BNT * Dm);        // vals = nbr+PE
    addpe(st, T0, XF, nullptr, Tt, Dm, (long long)BT * Dm);         // xb = x+PE
    ln(st, T0, P[98], P[99], T1, BT, Dm);                           // xn
    lin(st, T1, Dm, P[100], P[101], T2, Dm, BT, Dm, Dm, 0);         // Q
    lin(st, G3, Dm, P[102], P[103], G0, Dm, BNT, Dm, Dm, 0);        // Kbuf (B,2048,256)
    lin(st, G1, Dm, P[104], P[105], G2, Dm, BNT, Dm, Dm, 0);        // Vbuf
    gemm(st, T2, (long long)Tt * Dm, dk, Dm,
             G0, (long long)LK * Dm, dk, Dm, 1,
             nullptr, nullptr, 0, 0,
             S, (long long)HN * Tt * LK, (long long)Tt * LK, LK,
             Tt, LK, dk, Bn * HN, HN, 0);
    k_softmax<<<dim3(Tt, Bn * HN), dim3(256), 0, st>>>(S, Tt, LK, 1.f / sqrtf((float)dk),
                                                       amask, Nn, Tt, HN);
    gemm(st, S, (long long)HN * Tt * LK, (long long)Tt * LK, LK,
             G2, (long long)LK * Dm, dk, Dm, 0,
             nullptr, nullptr, 0, 0,
             G1, (long long)Tt * Dm, dk, Dm,
             Tt, dk, LK, Bn * HN, HN, 0);
    lin(st, G1, Dm, P[106], P[107], T0, Dm, BT, Dm, Dm, 0, T0);     // out proj + xb
    lin(st, T0, Dm, P[126], P[127], T1, 2 * FD, BT, 2 * FD, Dm, 0); // pred1
    lin(st, T1, 2 * FD, P[128], P[129], T2, 3, BT, 3, 2 * FD, 0);   // pred2 -> (B,T,3)
    long long tot = (long long)Bn * 3 * Tt;
    k_btc2bct<<<dim3((unsigned)((tot + 255) / 256)), dim3(256), 0, st>>>(T2, out, Tt, 3, tot);
  }

  // =================== precip neighbor-attention branch ===================
  {
    const int d = PEMB, dk = PEMB / HN;         // 64, 16
    lin(st, XF, Dm, P[84], P[85], XN, d, BT, d, Dm, 0);             // xp -> XN
    bct2btc(st, pm_in, T0, 1, Tt, 0, 17, (long long)Bn * Nn * 1 * Tt);
    bct2btc(st, pn_in, T0, FAUX, Tt, 1, 17, (long long)BNT * FAUX);
    lin(st, T0, 17, P[108], P[109], T1, d, BNT, d, 17, 1);          // pj1 + relu
    lin(st, T1, d, P[110], P[111], T2, d, BNT, d, d, 0);            // pj2 -> pnbr (T2)
    bct2btc(st, pe_in, T0, EAC, Tt, 0, EAC, (long long)BNT * EAC);
    lin(st, T0, EAC, P[112], P[113], T1, d, BNT, d, EAC, 1);        // ea1 + relu
    lin(st, T1, d, P[114], P[115], G0, d, BNT, d, d, 0);            // ea2 -> edge (G0)
    addpe(st, G3, T2, G0, Tt, d, (long long)BNT * d);               // keys
    addpe(st, G1, T2, nullptr, Tt, d, (long long)BNT * d);          // vals
    addpe(st, XN, XN, nullptr, Tt, d, (long long)BT * d);           // xb = xp+PE
    ln(st, XN, P[116], P[117], T0, BT, d);                          // xn
    lin(st, T0, d, P[118], P[119], T1, d, BT, d, d, 0);             // Q
    lin(st, G3, d, P[120], P[121], G2, d, BNT, d, d, 0);            // Kbuf (B,2048,64)
    lin(st, G1, d, P[122], P[123], G0, d, BNT, d, d, 0);            // Vbuf
    gemm(st, T1, (long long)Tt * d, dk, d,
             G2, (long long)LK * d, dk, d, 1,
             nullptr, nullptr, 0, 0,
             S, (long long)HN * Tt * LK, (long long)Tt * LK, LK,
             Tt, LK, dk, Bn * HN, HN, 0);
    k_softmax<<<dim3(Tt, Bn * HN), dim3(256), 0, st>>>(S, Tt, LK, 1.f / sqrtf((float)dk),
                                                       pmask, Nn, Tt, HN);
    gemm(st, S, (long long)HN * Tt * LK, (long long)Tt * LK, LK,
             G0, (long long)LK * d, dk, d, 0,
             nullptr, nullptr, 0, 0,
             T2, (long long)Tt * d, dk, d,
             Tt, dk, LK, Bn * HN, HN, 0);
    lin(st, T2, d, P[124], P[125], XN, d, BT, d, d, 0, XN);         // out proj + xb
    lin(st, XN, d, P[130], P[131], T0, d / 2, BT, d / 2, d, 0);     // pp1
    lin(st, T0, d / 2, P[132], P[133], out + (long long)Bn * 3 * Tt,
        1, BT, 1, d / 2, 0);                                        // pp2 -> (B,T,1)==(B,1,T)
  }
}
